// LSTM_824633721296
// MI455X (gfx1250) — matrix operand order repaired and checked by reading, not compiled, not run on hardware
//
#include <hip/hip_runtime.h>

typedef __attribute__((ext_vector_type(16))) _Float16 v16h;
typedef __attribute__((ext_vector_type(8)))  _Float16 v8h;
typedef __attribute__((ext_vector_type(8)))  float    v8f;
typedef __attribute__((ext_vector_type(4)))  float    v4f;

#define B_  1024
#define T_  512
#define I_  8
#define H_  64
#define G_  256   // 4*H
#define O_  10

// ---- fast activations: use CDNA5 v_tanh_f32 when the builtin exists ----
#if __has_builtin(__builtin_amdgcn_tanhf)
__device__ __forceinline__ float tanh_fast(float x) {
  return __builtin_amdgcn_tanhf(x);
}
#else
__device__ __forceinline__ float tanh_fast(float x) {
  // rcp-based fast path; avoids IEEE v_div_scale sequences
  float e = __expf(2.0f * x);
  return 1.0f - __fdividef(2.0f, e + 1.0f);
}
#endif

__device__ __forceinline__ float sigmoidf_(float x) {
  // sigmoid(x) = 0.5*tanh(0.5x) + 0.5  -> mul + tanh + fma
  return __builtin_fmaf(0.5f, tanh_fast(0.5f * x), 0.5f);
}

// Load 16 contiguous f32 weights (row of W, K-range) and convert to f16 B-operand chunk.
__device__ __forceinline__ v16h load_w16(const float* __restrict__ p) {
  v4f a = *(const v4f*)(p);
  v4f b = *(const v4f*)(p + 4);
  v4f c = *(const v4f*)(p + 8);
  v4f d = *(const v4f*)(p + 12);
  v16h r;
#pragma unroll
  for (int i = 0; i < 4; ++i) {
    r[i]      = (_Float16)a[i];
    r[i + 4]  = (_Float16)b[i];
    r[i + 8]  = (_Float16)c[i];
    r[i + 12] = (_Float16)d[i];
  }
  return r;
}

// Build the f16 16x32 A-operand from an LDS-resident 16x64 f16 h-state.
// Layout: lane<16 -> row=lane, K = kb+0..15 (elems 0..15)
//         lane>=16 -> row=lane-16, K = kb+16..31
__device__ __forceinline__ v16h lds_loadA(const _Float16* h, int lane, int kb) {
  int row  = lane & 15;
  int koff = kb + ((lane >> 4) << 4);
  v8h lo = *(const v8h*)(h + row * H_ + koff);
  v8h hi = *(const v8h*)(h + row * H_ + koff + 8);
  return __builtin_shufflevector(lo, hi, 0,1,2,3,4,5,6,7,8,9,10,11,12,13,14,15);
}

__global__ __launch_bounds__(512)
void lstm_fused_kernel(const float* __restrict__ x,
                       const float* __restrict__ w_ih0, const float* __restrict__ w_hh0,
                       const float* __restrict__ b_ih0, const float* __restrict__ b_hh0,
                       const float* __restrict__ w_ih1, const float* __restrict__ w_hh1,
                       const float* __restrict__ b_ih1, const float* __restrict__ b_hh1,
                       const float* __restrict__ w_fc,  const float* __restrict__ b_fc,
                       float* __restrict__ out) {
  __shared__ float    gatesT[G_ * 16];   // [gate_col][m], 16 KB, reused by both layers
  __shared__ _Float16 h0s[16 * H_];      // layer-0 hidden, f16, 2 KB
  __shared__ _Float16 h1s[16 * H_];      // layer-1 hidden, f16, 2 KB
  __shared__ float    h1f[16 * H_];      // layer-1 hidden f32 at last step, 4 KB

  const int tid   = threadIdx.x;
  const int lane  = tid & 31;
  const int wv    = tid >> 5;                 // wave id 0..15 == N-tile
  const int b0    = blockIdx.x * 16;          // batch tile base
  const int colg  = wv * 16 + (lane & 15);    // gate column owned by this lane (B operand)
  const int khalf = (lane >> 4);              // 0 or 1 (K-half of B/A layouts)

  // ---------------- preload weights as f16 B-operands (persist in VGPRs) ----------------
  v16h Bih0;                                   // K padded 8 -> 32
  {
    const float* p = w_ih0 + colg * I_;
#pragma unroll
    for (int i = 0; i < 8; ++i)
      Bih0[i] = (khalf == 0) ? (_Float16)p[i] : (_Float16)0.f;
#pragma unroll
    for (int i = 8; i < 16; ++i)
      Bih0[i] = (_Float16)0.f;
  }
  const v16h Bhh0_0 = load_w16(w_hh0 + colg * H_ +      khalf * 16);
  const v16h Bhh0_1 = load_w16(w_hh0 + colg * H_ + 32 + khalf * 16);
  const v16h Bih1_0 = load_w16(w_ih1 + colg * H_ +      khalf * 16);
  const v16h Bih1_1 = load_w16(w_ih1 + colg * H_ + 32 + khalf * 16);
  const v16h Bhh1_0 = load_w16(w_hh1 + colg * H_ +      khalf * 16);
  const v16h Bhh1_1 = load_w16(w_hh1 + colg * H_ + 32 + khalf * 16);

  // ---------------- cell-update thread state ----------------
  const int uj = tid >> 3;   // hidden index 0..63
  const int um = tid & 7;    // rows um and um+8
  float bias0[4], bias1[4];
#pragma unroll
  for (int q = 0; q < 4; ++q) {
    bias0[q] = b_ih0[uj + 64 * q] + b_hh0[uj + 64 * q];
    bias1[q] = b_ih1[uj + 64 * q] + b_hh1[uj + 64 * q];
  }
  float c0[2] = {0.f, 0.f};
  float c1[2] = {0.f, 0.f};

  // zero-init h state (1024 halves each, 512 threads x2)
  h0s[tid] = (_Float16)0.f; h0s[tid + 512] = (_Float16)0.f;
  h1s[tid] = (_Float16)0.f; h1s[tid + 512] = (_Float16)0.f;
  __syncthreads();

  const float* xrow = x + (size_t)(b0 + (lane & 15)) * T_ * I_;

#pragma unroll 1
  for (int t = 0; t < T_; ++t) {
    // ---- A operand for x_t (K=8, padded to 32; lanes>=16 are the zero K-halves) ----
    v4f x0 = *(const v4f*)(xrow + (size_t)t * I_);
    v4f x1 = *(const v4f*)(xrow + (size_t)t * I_ + 4);
    v16h Ax;
#pragma unroll
    for (int i = 0; i < 4; ++i) {
      Ax[i]      = (khalf == 0) ? (_Float16)x0[i] : (_Float16)0.f;
      Ax[i + 4]  = (khalf == 0) ? (_Float16)x1[i] : (_Float16)0.f;
      Ax[i + 8]  = (_Float16)0.f;
      Ax[i + 12] = (_Float16)0.f;
    }

    // ---- layer 0 gates: x_t @ Wih0^T + h0 @ Whh0^T ----
    v8f acc0 = {0.f,0.f,0.f,0.f,0.f,0.f,0.f,0.f};
    acc0 = __builtin_amdgcn_wmma_f32_16x16x32_f16(false, Ax, false, Bih0, (short)0, acc0, false, false);
    {
      v16h a0 = lds_loadA(h0s, lane, 0);
      v16h a1 = lds_loadA(h0s, lane, 32);
      acc0 = __builtin_amdgcn_wmma_f32_16x16x32_f16(false, a0, false, Bhh0_0, (short)0, acc0, false, false);
      acc0 = __builtin_amdgcn_wmma_f32_16x16x32_f16(false, a1, false, Bhh0_1, (short)0, acc0, false, false);
    }
    {
      float* gp = &gatesT[colg * 16 + khalf * 8];
      *(v4f*)gp       = __builtin_shufflevector(acc0, acc0, 0, 1, 2, 3);
      *(v4f*)(gp + 4) = __builtin_shufflevector(acc0, acc0, 4, 5, 6, 7);
    }
    __syncthreads();

    // ---- layer 0 cell update ----
#pragma unroll
    for (int s = 0; s < 2; ++s) {
      int m = um + 8 * s;
      float gi = gatesT[(uj      ) * 16 + m] + bias0[0];
      float gf = gatesT[(uj +  64) * 16 + m] + bias0[1];
      float gg = gatesT[(uj + 128) * 16 + m] + bias0[2];
      float go = gatesT[(uj + 192) * 16 + m] + bias0[3];
      float i_ = sigmoidf_(gi), f_ = sigmoidf_(gf);
      float g_ = tanh_fast(gg), o_ = sigmoidf_(go);
      c0[s] = f_ * c0[s] + i_ * g_;
      float h = o_ * tanh_fast(c0[s]);
      h0s[m * H_ + uj] = (_Float16)h;
    }
    __syncthreads();

    // ---- layer 1 gates: h0 @ Wih1^T + h1 @ Whh1^T ----
    v8f acc1 = {0.f,0.f,0.f,0.f,0.f,0.f,0.f,0.f};
    {
      v16h a0 = lds_loadA(h0s, lane, 0);
      v16h a1 = lds_loadA(h0s, lane, 32);
      acc1 = __builtin_amdgcn_wmma_f32_16x16x32_f16(false, a0, false, Bih1_0, (short)0, acc1, false, false);
      acc1 = __builtin_amdgcn_wmma_f32_16x16x32_f16(false, a1, false, Bih1_1, (short)0, acc1, false, false);
      v16h a2 = lds_loadA(h1s, lane, 0);
      v16h a3 = lds_loadA(h1s, lane, 32);
      acc1 = __builtin_amdgcn_wmma_f32_16x16x32_f16(false, a2, false, Bhh1_0, (short)0, acc1, false, false);
      acc1 = __builtin_amdgcn_wmma_f32_16x16x32_f16(false, a3, false, Bhh1_1, (short)0, acc1, false, false);
    }
    {
      float* gp = &gatesT[colg * 16 + khalf * 8];
      *(v4f*)gp       = __builtin_shufflevector(acc1, acc1, 0, 1, 2, 3);
      *(v4f*)(gp + 4) = __builtin_shufflevector(acc1, acc1, 4, 5, 6, 7);
    }
    __syncthreads();

    // ---- layer 1 cell update ----
#pragma unroll
    for (int s = 0; s < 2; ++s) {
      int m = um + 8 * s;
      float gi = gatesT[(uj      ) * 16 + m] + bias1[0];
      float gf = gatesT[(uj +  64) * 16 + m] + bias1[1];
      float gg = gatesT[(uj + 128) * 16 + m] + bias1[2];
      float go = gatesT[(uj + 192) * 16 + m] + bias1[3];
      float i_ = sigmoidf_(gi), f_ = sigmoidf_(gf);
      float g_ = tanh_fast(gg), o_ = sigmoidf_(go);
      c1[s] = f_ * c1[s] + i_ * g_;
      float h = o_ * tanh_fast(c1[s]);
      h1s[m * H_ + uj] = (_Float16)h;
      if (t == T_ - 1) h1f[m * H_ + uj] = h;
    }
    __syncthreads();
  }

  // ---- final FC: out[b0+m][o] = h1_last[m] . w_fc[o] + b_fc[o] ----
  if (tid < 16 * O_) {
    int m = tid / O_;
    int o = tid % O_;
    const float* wr = w_fc + o * H_;
    float s = b_fc[o];
#pragma unroll
    for (int k = 0; k < H_; ++k) s += h1f[m * H_ + k] * wr[k];
    out[(b0 + m) * O_ + o] = s;
  }
}

extern "C" void kernel_launch(void* const* d_in, const int* in_sizes, int n_in,
                              void* d_out, int out_size, void* d_ws, size_t ws_size,
                              hipStream_t stream) {
  (void)in_sizes; (void)n_in; (void)out_size; (void)d_ws; (void)ws_size;
  const float* x     = (const float*)d_in[0];
  const float* w_ih0 = (const float*)d_in[1];
  const float* w_hh0 = (const float*)d_in[2];
  const float* b_ih0 = (const float*)d_in[3];
  const float* b_hh0 = (const float*)d_in[4];
  const float* w_ih1 = (const float*)d_in[5];
  const float* w_hh1 = (const float*)d_in[6];
  const float* b_ih1 = (const float*)d_in[7];
  const float* b_hh1 = (const float*)d_in[8];
  const float* w_fc  = (const float*)d_in[9];
  const float* b_fc  = (const float*)d_in[10];
  float* out = (float*)d_out;

  dim3 grid(B_ / 16);   // 64 workgroups, one 16-row batch tile each
  dim3 block(512);      // 16 waves: one per 16-column gate tile
  lstm_fused_kernel<<<grid, block, 0, stream>>>(x, w_ih0, w_hh0, b_ih0, b_hh0,
                                                w_ih1, w_hh1, b_ih1, b_hh1,
                                                w_fc, b_fc, out);
}
